// DampConv2d_14491219656887
// MI455X (gfx1250) — matrix operand order repaired and checked by reading, not compiled, not run on hardware
//
#include <hip/hip_runtime.h>
#include <hip/hip_bf16.h>

// DampConv2d on MI455X (gfx1250): implicit-GEMM 3x3 conv via
// V_WMMA_F32_16X16X32_BF16 (fp32 accumulate).
// - 8-wave workgroup; each wave owns 16 output channels with its A operand
//   (weights) register-resident for the whole block (18 x v16bf = 144 VGPRs).
// - Shared LDS x-halo tile (damp mask fused at staging), 64 t per block,
//   B fragments via conflict-free ds_load_b128.

typedef __bf16 bf16_t;
typedef __attribute__((ext_vector_type(16))) __bf16 v16bf;
typedef __attribute__((ext_vector_type(8)))  __bf16 v8bf;
typedef __attribute__((ext_vector_type(8)))  float  v8f;

#define N_BATCH 32
#define C_IN    64
#define C_OUT   128
#define F_DIM   128
#define T_DIM   256
#define TAPS    9
#define T_TILE  64                 // t-span per block (4 WMMA sub-tiles of 16)
#define T_HALO  (T_TILE + 2)       // 66
#define LDS_PITCH 72               // halves/row: 144B -> 36-bank stride, 16B aligned

// Faithful to reference: BOTH mask terms are functions of t.
__device__ __forceinline__ float damp_mask(int t) {
  float tf = (float)t;
  float term1 = 1.0f - 0.1f * fabsf(tf - (float)(T_DIM / 2)) / (float)(T_DIM / 2);
  float term2 = 1.0f - 0.1f * fabsf(tf - (float)(F_DIM / 2)) / (float)(F_DIM / 2);
  return term1 * term2;
}

// Repack W[o][c][kh][kw] fp32 -> Wp[o][tap][c] bf16 (tap = kh*3+kw) so
// per-lane A fragments are two contiguous 16B loads.
__global__ __launch_bounds__(256) void pack_weights_kernel(
    const float* __restrict__ W, bf16_t* __restrict__ Wp) {
  int idx = blockIdx.x * blockDim.x + threadIdx.x;
  if (idx >= C_OUT * TAPS * C_IN) return;
  int c   = idx % C_IN;
  int tap = (idx / C_IN) % TAPS;
  int o   = idx / (C_IN * TAPS);
  Wp[idx] = (bf16_t)W[(o * C_IN + c) * TAPS + tap];
}

// Block = (t-block of 64, f, n); 8 waves, wave w computes channels [16w,16w+16).
__global__ __launch_bounds__(256) void damp_conv_wmma_kernel(
    const float*  __restrict__ x,    // [N][C][F][T]
    const bf16_t* __restrict__ Wp,   // [O][tap][C] bf16
    const float*  __restrict__ bias, // [O]
    float*        __restrict__ out)  // [N][O][F][T]
{
  __shared__ bf16_t lds[3 * T_HALO * LDS_PITCH];   // 28512 B

  const int t0   = blockIdx.x * T_TILE;
  const int f    = blockIdx.y;
  const int n    = blockIdx.z;
  const int tid  = threadIdx.x;
  const int wave = tid >> 5;          // 0..7 -> output-channel block
  const int lane = tid & 31;

  // ---- WMMA lane geometry (CDNA5 16-bit layouts) ----
  const bool loHalf = lane < 16;
  const int  Nn     = lane & 15;      // A: row (o); B/D: column (t)
  const int  cA     = loHalf ? 0 : 8; // A lane-group K base
  const int  cB     = loHalf ? 0 : 8; // B lane-group K base

  // ---- Load this wave's A operand once: 18 fragments, register-resident ----
  // afrag[tap][kc]: halves j=0..7 -> c0+cA+j ; j=8..15 -> c0+cA+16+(j-8).
  v16bf afrag[TAPS][2];
  #pragma unroll
  for (int tap = 0; tap < TAPS; ++tap) {
    #pragma unroll
    for (int kc = 0; kc < 2; ++kc) {
      const int o = wave * 16 + Nn;
      const bf16_t* wrow = Wp + (o * TAPS + tap) * C_IN + kc * 32 + cA;
      v8bf alo = *(const v8bf*)(wrow);        // global_load_b128
      v8bf ahi = *(const v8bf*)(wrow + 16);   // global_load_b128
      #pragma unroll
      for (int i = 0; i < 8; ++i) {
        afrag[tap][kc][i]     = alo[i];
        afrag[tap][kc][i + 8] = ahi[i];
      }
    }
  }

  // ---- Cooperatively stage damped input halo tile: [fr][tl][c] ----
  for (int e = tid; e < 3 * T_HALO * C_IN; e += 256) {
    int fr  = e / (T_HALO * C_IN);
    int rem = e - fr * (T_HALO * C_IN);
    int c   = rem / T_HALO;
    int tl  = rem % T_HALO;            // lanes sweep t fastest -> coalesced
    int fi  = f + fr - 1;
    int ti  = t0 + tl - 1;
    float v = 0.0f;
    if ((unsigned)fi < (unsigned)F_DIM && (unsigned)ti < (unsigned)T_DIM)
      v = x[((n * C_IN + c) * F_DIM + fi) * T_DIM + ti] * damp_mask(ti);
    lds[(fr * T_HALO + tl) * LDS_PITCH + c] = (bf16_t)v;
  }
  __syncthreads();

  // ---- Per-lane bias values for the 8 D rows this lane stores ----
  const int obase = wave * 16 + (loHalf ? 0 : 8);
  float bi[8];
  #pragma unroll
  for (int r = 0; r < 8; ++r) bi[r] = bias[obase + r];

  // ---- Compute 4 sub-tiles of 16 t; A from registers, B from LDS ----
  for (int st = 0; st < 4; ++st) {
    v8f acc = (v8f){0.f, 0.f, 0.f, 0.f, 0.f, 0.f, 0.f, 0.f};

    #pragma unroll
    for (int tap = 0; tap < TAPS; ++tap) {
      const int kh = tap / 3;
      const int kw = tap % 3;
      const bf16_t* ldsrow =
          &lds[(kh * T_HALO + (st * 16 + Nn + kw)) * LDS_PITCH];

      #pragma unroll
      for (int kc = 0; kc < 2; ++kc) {
        // B fragment (32x16 K x t): halves j=0..7 -> cB+j ; j=8..15 -> cB+16+(j-8).
        v8bf blo = *(const v8bf*)(ldsrow + kc * 32 + cB);      // ds_load_b128
        v8bf bhi = *(const v8bf*)(ldsrow + kc * 32 + cB + 16); // ds_load_b128
        v16bf bfrag;
        #pragma unroll
        for (int i = 0; i < 8; ++i) { bfrag[i] = blo[i]; bfrag[i + 8] = bhi[i]; }

        acc = __builtin_amdgcn_wmma_f32_16x16x32_bf16(
            /*neg_a=*/false, afrag[tap][kc], /*neg_b=*/false, bfrag,
            /*c_mod=*/(short)0, acc, /*reuse_a=*/false, /*reuse_b=*/false);
      }
    }

    // ---- Store D: lane L, vgpr r -> o = obase + r, t = t0 + st*16 + L%16
    const int tglob = t0 + st * 16 + Nn;
    #pragma unroll
    for (int r = 0; r < 8; ++r) {
      out[((n * C_OUT + obase + r) * F_DIM + f) * T_DIM + tglob] = acc[r] + bi[r];
    }
  }
}

extern "C" void kernel_launch(void* const* d_in, const int* in_sizes, int n_in,
                              void* d_out, int out_size, void* d_ws, size_t ws_size,
                              hipStream_t stream) {
  const float* x = (const float*)d_in[0];   // [32,64,128,256]
  const float* W = (const float*)d_in[1];   // [128,64,3,3]
  const float* b = (const float*)d_in[2];   // [128]
  float* out     = (float*)d_out;           // [32,128,128,256]
  bf16_t* Wp     = (bf16_t*)d_ws;           // 128*9*64 bf16 = 147456 B

  (void)in_sizes; (void)n_in; (void)out_size; (void)ws_size;

  const int wElems = C_OUT * TAPS * C_IN;
  pack_weights_kernel<<<(wElems + 255) / 256, 256, 0, stream>>>(W, Wp);

  dim3 grid(T_DIM / T_TILE, F_DIM, N_BATCH);   // (4, 128, 32)
  damp_conv_wmma_kernel<<<grid, 256, 0, stream>>>(x, Wp, b, out);
}
